// RadDet_61744449847384
// MI455X (gfx1250) — matrix operand order repaired and checked by reading, not compiled, not run on hardware
//
#include <hip/hip_runtime.h>
#include <hip/hip_bf16.h>
#include <math.h>

typedef __attribute__((ext_vector_type(16))) __bf16 v16bf;
typedef __attribute__((ext_vector_type(8)))  float  v8f;

#define SEQ   4096
#define EMBED 256

// ---------------------------------------------------------------------------
// Patch gather: x (B,1,64,64,64) -> xt (B, 4096, 64)
// ---------------------------------------------------------------------------
__global__ void patch_gather_k(const float* __restrict__ x, float* __restrict__ xt, int B) {
  int idx = blockIdx.x * blockDim.x + threadIdx.x;
  if (idx >= B * SEQ * 64) return;
  int e   = idx & 63;
  int tok = (idx >> 6) & (SEQ - 1);
  int b   = idx >> 18;                 // 64 * 4096 = 2^18
  int pd = e & 3, pw = (e >> 2) & 3, ph = e >> 4;
  int dd = tok & 15, ww = (tok >> 4) & 15, hh = tok >> 8;
  int X = hh * 4 + ph, Y = ww * 4 + pw, Z = dd * 4 + pd;
  xt[idx] = x[(((size_t)b * 64 + X) * 64 + Y) * 64 + Z];
}

// ---------------------------------------------------------------------------
// Register-blocked GEMM: C[M,N] = A[M,K] @ W^T (+bias) (+C).
// Each wave computes an MT x NT grid of 16x16 tiles via
// v_wmma_f32_16x16x32_bf16: per k-step MT+NT fragment loads feed MT*NT WMMAs.
// K / HASBIAS / ADDC are compile-time -> no runtime branches anywhere.
//   WTRANS=false: W is [N,K] row-contiguous (wstride = K)
//   WTRANS=true : W is [K,N] (wstride = N), strided loads (patch GEMM only)
// ---------------------------------------------------------------------------
template<int K, int MT, int NT, bool WTRANS, bool HASBIAS, bool ADDC>
__global__ void gemm_wmma_bf16(const float* __restrict__ A, int lda,
                               const float* __restrict__ W, int wstride,
                               const float* __restrict__ bias,
                               float* __restrict__ C, int ldc,
                               int M, int N) {
  int wave    = (blockIdx.x * blockDim.x + threadIdx.x) >> 5;
  int lane    = threadIdx.x & 31;
  int nblocks = N / (16 * NT);
  int mblocks = M / (16 * MT);
  int bm = wave / nblocks;
  int bn = wave - bm * nblocks;
  if (bm >= mblocks) return;            // wave-uniform exit (EXEC stays all-1)

  int l15   = lane & 15;
  int khalf = lane >> 4;

  const float* arow[MT];
#pragma unroll
  for (int mi = 0; mi < MT; ++mi)
    arow[mi] = A + (size_t)((bm * MT + mi) * 16 + l15) * lda;
  const float* wcol[NT];
#pragma unroll
  for (int ni = 0; ni < NT; ++ni)
    wcol[ni] = W + (size_t)((bn * NT + ni) * 16 + l15) * (WTRANS ? 1 : wstride);

  v8f acc[MT][NT];
#pragma unroll
  for (int mi = 0; mi < MT; ++mi)
#pragma unroll
    for (int ni = 0; ni < NT; ++ni) acc[mi][ni] = (v8f){};

  for (int k0 = 0; k0 < K; k0 += 32) {
    v16bf a[MT], b[NT];
    // ---- A fragments: two contiguous 8-float runs per lane (ISA 16-bit A 16x32)
#pragma unroll
    for (int mi = 0; mi < MT; ++mi) {
#pragma unroll
      for (int g = 0; g < 2; ++g) {
        if (k0 + g * 16 + 16 <= K) {    // folds at compile time
          const float4* p = (const float4*)(arow[mi] + k0 + g * 16 + khalf * 8);
          float4 f0 = p[0], f1 = p[1];
          a[mi][g * 8 + 0] = (__bf16)f0.x; a[mi][g * 8 + 1] = (__bf16)f0.y;
          a[mi][g * 8 + 2] = (__bf16)f0.z; a[mi][g * 8 + 3] = (__bf16)f0.w;
          a[mi][g * 8 + 4] = (__bf16)f1.x; a[mi][g * 8 + 5] = (__bf16)f1.y;
          a[mi][g * 8 + 6] = (__bf16)f1.z; a[mi][g * 8 + 7] = (__bf16)f1.w;
        } else {
#pragma unroll
          for (int j = 0; j < 8; ++j) a[mi][g * 8 + j] = (__bf16)0.0f;
        }
      }
    }
    // ---- B fragments: two contiguous 8-float runs per lane (ISA 16-bit B 32x16)
    int bbase = k0 + khalf * 8;
#pragma unroll
    for (int ni = 0; ni < NT; ++ni) {
      if (!WTRANS) {
        if (k0 + 32 <= K) {             // folds at compile time
          const float4* p = (const float4*)(wcol[ni] + bbase);
          float4 f0 = p[0], f1 = p[1], f2 = p[4], f3 = p[5];
          b[ni][0]  = (__bf16)f0.x; b[ni][1]  = (__bf16)f0.y;
          b[ni][2]  = (__bf16)f0.z; b[ni][3]  = (__bf16)f0.w;
          b[ni][4]  = (__bf16)f1.x; b[ni][5]  = (__bf16)f1.y;
          b[ni][6]  = (__bf16)f1.z; b[ni][7]  = (__bf16)f1.w;
          b[ni][8]  = (__bf16)f2.x; b[ni][9]  = (__bf16)f2.y;
          b[ni][10] = (__bf16)f2.z; b[ni][11] = (__bf16)f2.w;
          b[ni][12] = (__bf16)f3.x; b[ni][13] = (__bf16)f3.y;
          b[ni][14] = (__bf16)f3.z; b[ni][15] = (__bf16)f3.w;
        } else {                        // only K=16 instantiation
#pragma unroll
          for (int e = 0; e < 16; ++e)
            b[ni][e] = (bbase + e + (e >> 3) * 8 < K) ? (__bf16)wcol[ni][bbase + e + (e >> 3) * 8] : (__bf16)0.0f;
        }
      } else {                          // [K,N] weights: strided (patch GEMM only)
#pragma unroll
        for (int e = 0; e < 16; ++e)
          b[ni][e] = (__bf16)wcol[ni][(size_t)(bbase + e + (e >> 3) * 8) * wstride];
      }
    }
    // ---- MT*NT WMMAs per k-step
#pragma unroll
    for (int mi = 0; mi < MT; ++mi)
#pragma unroll
      for (int ni = 0; ni < NT; ++ni)
        acc[mi][ni] = __builtin_amdgcn_wmma_f32_16x16x32_bf16(
            false, a[mi], false, b[ni], (short)0, acc[mi][ni], false, false);
  }

  // ---- epilogue: straight-line (HASBIAS/ADDC compile-time)
#pragma unroll
  for (int mi = 0; mi < MT; ++mi) {
#pragma unroll
    for (int ni = 0; ni < NT; ++ni) {
      int cn = (bn * NT + ni) * 16 + l15;
      int mb = (bm * MT + mi) * 16 + khalf * 8;
      float bv = HASBIAS ? bias[cn] : 0.0f;
#pragma unroll
      for (int r = 0; r < 8; ++r) {
        size_t off = (size_t)(mb + r) * ldc + cn;
        float v = acc[mi][ni][r] + bv;
        if (ADDC) v += C[off];
        C[off] = v;
      }
    }
  }
}

// ---------------------------------------------------------------------------
// Physical positional embedding (normalization cancels resolutions: c = idx/15)
// channels: [0,86) sin/cos(h/15), [86,171) sin/cos(w/15)+lin, [171,256) d/15
// ---------------------------------------------------------------------------
__global__ void posembed_add_k(float* __restrict__ tok, int B) {
  int idx = blockIdx.x * blockDim.x + threadIdx.x;
  if (idx >= B * SEQ * EMBED) return;
  int ch = idx & 255;
  int t  = (idx >> 8) & (SEQ - 1);
  int hh = t >> 8, ww = (t >> 4) & 15, dd = t & 15;
  float val;
  if (ch < 86) {
    float c = (float)hh * (1.0f / 15.0f);
    int  k   = ch;
    bool isc = (k >= 43); if (isc) k -= 43;
    float om = powf(10000.0f, -(float)k * (1.0f / 43.0f));
    float o  = c * om;
    val = isc ? cosf(o) : sinf(o);
  } else if (ch < 171) {
    int cc = ch - 86;
    float c = (float)ww * (1.0f / 15.0f);
    if (cc < 84) {
      bool isc = (cc >= 42); int k = isc ? cc - 42 : cc;
      float om = powf(10000.0f, -(float)k * (1.0f / 42.0f));
      float o  = c * om;
      val = isc ? cosf(o) : sinf(o);
    } else val = c * 0.01f;
  } else {
    int cc = ch - 171;
    float c = (float)dd * (1.0f / 15.0f);
    if (cc < 84) {
      bool isc = (cc >= 42); int k = isc ? cc - 42 : cc;
      float om = powf(10000.0f, -(float)k * (1.0f / 42.0f));
      float o  = c * om;
      val = isc ? cosf(o) : sinf(o);
    } else val = c * 0.01f;
  }
  tok[idx] += val;
}

// ---------------------------------------------------------------------------
// LayerNorm over d=256, one block (256 threads) per row
// ---------------------------------------------------------------------------
__global__ void layernorm_k(const float* __restrict__ x, const float* __restrict__ w,
                            const float* __restrict__ b, float* __restrict__ y) {
  __shared__ float red[256];
  int row = blockIdx.x, tid = threadIdx.x;
  float v = x[(size_t)row * EMBED + tid];
  red[tid] = v; __syncthreads();
  for (int s = 128; s > 0; s >>= 1) { if (tid < s) red[tid] += red[tid + s]; __syncthreads(); }
  float mu = red[0] * (1.0f / EMBED);
  __syncthreads();
  float dv = v - mu;
  red[tid] = dv * dv; __syncthreads();
  for (int s = 128; s > 0; s >>= 1) { if (tid < s) red[tid] += red[tid + s]; __syncthreads(); }
  float var = red[0] * (1.0f / EMBED);
  y[(size_t)row * EMBED + tid] = dv * rsqrtf(var + 1e-5f) * w[tid] + b[tid];
}

// ---------------------------------------------------------------------------
// Causal depthwise conv (K=4) on x_in = xz[:, :256] (row stride 512) + SiLU
// ---------------------------------------------------------------------------
__global__ void dwconv_silu_k(const float* __restrict__ xz, const float* __restrict__ cw,
                              const float* __restrict__ cb, float* __restrict__ xc, int B) {
  int idx = blockIdx.x * blockDim.x + threadIdx.x;
  if (idx >= B * SEQ * EMBED) return;
  int c = idx & 255;
  int l = (idx >> 8) & (SEQ - 1);
  int b = idx >> 20;                     // 256 * 4096 = 2^20
  const float* base = xz + (size_t)b * SEQ * 512 + c;
  float s = cb[c];
#pragma unroll
  for (int k = 0; k < 4; ++k) {
    int ls = l + k - 3;
    if (ls >= 0) s += cw[c * 4 + k] * base[(size_t)ls * 512];
  }
  float sg = 1.0f / (1.0f + expf(-s));
  xc[idx] = s * sg;
}

__global__ void softplus_k(float* __restrict__ d, int n) {
  int idx = blockIdx.x * blockDim.x + threadIdx.x;
  if (idx >= n) return;
  float v = d[idx];
  d[idx] = (v > 20.0f) ? v : log1pf(expf(v));
}

// ---------------------------------------------------------------------------
// Selective scan. block (16,16): tx = state n, ty = channel-in-group.
// grid (16, B). 16-lane shfl_xor reduction over n (wave32: two c's per wave).
// ---------------------------------------------------------------------------
__global__ void scan_k(const float* __restrict__ delta, const float* __restrict__ xdbl,
                       const float* __restrict__ xc, const float* __restrict__ A_log,
                       const float* __restrict__ D_skip, float* __restrict__ y) {
  int n = threadIdx.x, cl = threadIdx.y;
  int b = blockIdx.y;
  int c = blockIdx.x * 16 + cl;
  float Ac = -expf(A_log[c * 16 + n]);
  float Ds = D_skip[c];
  float h  = 0.0f;
  size_t rowBase = (size_t)b * SEQ;
  for (int l = 0; l < SEQ; ++l) {
    size_t r = rowBase + l;
    if (l + 4 < SEQ) __builtin_prefetch(&xdbl[(r + 4) * 48], 0, 1);
    float dl = delta[r * 256 + c];
    float xv = xc[r * 256 + c];
    float Bn = xdbl[r * 48 + 16 + n];
    float Cn = xdbl[r * 48 + 32 + n];
    h = expf(dl * Ac) * h + dl * Bn * xv;
    float p = h * Cn;
    p += __shfl_xor(p, 1, 16);
    p += __shfl_xor(p, 2, 16);
    p += __shfl_xor(p, 4, 16);
    p += __shfl_xor(p, 8, 16);
    if (n == 0) y[r * 256 + c] = p + xv * Ds;
  }
}

// y *= silu(z), z = xz[:, 256:512]
__global__ void gate_k(float* __restrict__ y, const float* __restrict__ xz, int n) {
  int idx = blockIdx.x * blockDim.x + threadIdx.x;
  if (idx >= n) return;
  int c = idx & 255;
  int row = idx >> 8;
  float z  = xz[(size_t)row * 512 + 256 + c];
  float sg = 1.0f / (1.0f + expf(-z));
  y[idx] *= z * sg;
}

// ---------------------------------------------------------------------------
template<int K, int MT, int NT, bool WTRANS, bool HASBIAS, bool ADDC>
static void launch_gemm(const float* A, int lda, const float* W, int wstride,
                        const float* bias, float* C, int ldc, int M, int N,
                        hipStream_t stream) {
  int waves   = (M / (16 * MT)) * (N / (16 * NT));
  int threads = waves * 32;
  int blocks  = (threads + 255) / 256;
  gemm_wmma_bf16<K, MT, NT, WTRANS, HASBIAS, ADDC>
      <<<blocks, 256, 0, stream>>>(A, lda, W, wstride, bias, C, ldc, M, N);
}

extern "C" void kernel_launch(void* const* d_in, const int* in_sizes, int n_in,
                              void* d_out, int out_size, void* d_ws, size_t ws_size,
                              hipStream_t stream) {
  (void)n_in; (void)out_size; (void)ws_size;
  const float* x         = (const float*)d_in[0];
  // d_in[1..4]: range/angle/vel res + has_velocity -> cancel in normalization
  const float* patch_w   = (const float*)d_in[5];   // (64, 256)  [K,N]
  const float* patch_b   = (const float*)d_in[6];
  const float* ln_w      = (const float*)d_in[7];
  const float* ln_b      = (const float*)d_in[8];
  const float* in_proj_w = (const float*)d_in[9];   // (4, 512, 256) [N,K]
  const float* conv_w    = (const float*)d_in[10];  // (4, 256, 4)
  const float* conv_b    = (const float*)d_in[11];
  const float* xproj_w   = (const float*)d_in[12];  // (4, 48, 256)  [N,K]
  const float* dtproj_w  = (const float*)d_in[13];  // (4, 256, 16)  [N,K]
  const float* dtproj_b  = (const float*)d_in[14];
  const float* A_log     = (const float*)d_in[15];  // (4, 256, 16)
  const float* D_skip    = (const float*)d_in[16];
  const float* outproj_w = (const float*)d_in[17];  // (4, 256, 256) [N,K]
  const float* normf_w   = (const float*)d_in[18];
  const float* normf_b   = (const float*)d_in[19];

  int B    = in_sizes[0] / (64 * 64 * 64);
  int rows = B * SEQ;
  size_t rd = (size_t)rows * EMBED;

  // workspace carve-up (with aliasing): ~50 MB for B=2
  float* ws    = (float*)d_ws;
  float* hbuf  = ws;                          // residual     rows*256
  float* ubuf  = hbuf + rd;                   // LN out / y   rows*256 (aliased)
  float* yb    = ubuf;                        //   (u dead before scan writes y)
  float* xz    = ubuf + rd;                   // in_proj out  rows*512
  float* xtb   = xz;                          //   patch-gather aliases xz (dead after)
  float* xcb   = xz + (size_t)rows * 512;     // conv out     rows*256
  float* xdbl  = xcb + rd;                    // x_proj out   rows*48
  float* delta = xdbl + (size_t)rows * 48;    // softplus(dt) rows*256

  int eltot = rows * EMBED;
  int elblk = (eltot + 255) / 256;

  // --- patch embed + pos embed ---
  patch_gather_k<<<(B * SEQ * 64 + 255) / 256, 256, 0, stream>>>(x, xtb, B);
  launch_gemm<64, 4, 2, true, true, false>(xtb, 64, patch_w, /*wstride=N*/EMBED,
                                           patch_b, hbuf, EMBED, rows, EMBED, stream);
  posembed_add_k<<<elblk, 256, 0, stream>>>(hbuf, B);

  // --- 4 Mamba layers ---
  for (int i = 0; i < 4; ++i) {
    layernorm_k<<<rows, 256, 0, stream>>>(hbuf, ln_w + i * 256, ln_b + i * 256, ubuf);
    launch_gemm<256, 4, 2, false, false, false>(ubuf, 256, in_proj_w + (size_t)i * 512 * 256,
                                                256, nullptr, xz, 512, rows, 512, stream);
    dwconv_silu_k<<<elblk, 256, 0, stream>>>(xz, conv_w + i * 1024, conv_b + i * 256, xcb, B);
    launch_gemm<256, 4, 1, false, false, false>(xcb, 256, xproj_w + (size_t)i * 48 * 256,
                                                256, nullptr, xdbl, 48, rows, 48, stream);
    launch_gemm<16, 4, 2, false, true, false>(xdbl, 48, dtproj_w + (size_t)i * 256 * 16,
                                              16, dtproj_b + i * 256, delta, 256,
                                              rows, 256, stream);
    softplus_k<<<elblk, 256, 0, stream>>>(delta, eltot);
    scan_k<<<dim3(16, B), dim3(16, 16), 0, stream>>>(delta, xdbl, xcb,
                                                     A_log + i * 4096, D_skip + i * 256, yb);
    gate_k<<<elblk, 256, 0, stream>>>(yb, xz, eltot);
    launch_gemm<256, 4, 2, false, false, true>(yb, 256, outproj_w + (size_t)i * 256 * 256,
                                               256, nullptr, hbuf, 256, rows, 256, stream);
  }

  // --- final norm -> output ---
  layernorm_k<<<rows, 256, 0, stream>>>(hbuf, normf_w, normf_b, (float*)d_out);
}
